// MultiHeadAttentionWithRPE_23321672417445
// MI455X (gfx1250) — matrix operand order repaired and checked by reading, not compiled, not run on hardware
//
#include <hip/hip_runtime.h>
#include <hip/hip_bf16.h>

typedef __attribute__((ext_vector_type(16))) _Float16 v16h;
typedef __attribute__((ext_vector_type(8)))  _Float16 v8h;
typedef __attribute__((ext_vector_type(8)))  float    v8f;

#define WMMA_F16(a, b, c) \
  __builtin_amdgcn_wmma_f32_16x16x32_f16(false, (a), false, (b), (short)0, (c), false, false)

// A-matrix (16x32 f16) per-lane load, row-major source.
// Lane l: row m = l&15, ksel = 8*(l>>4), K = 2*ksel + 0..15 contiguous.
__device__ inline v16h load_a16(const _Float16* p, int ksel) {
  v16h a;
  *(v8h*)&a       = *(const v8h*)(p + 2 * ksel);
  *((v8h*)&a + 1) = *(const v8h*)(p + 2 * ksel + 8);
  return a;
}

// B-matrix (32x16 f16) per-lane load: lane l holds column n = l&15,
// K = 16*(l>>4) + 0..15, contiguous at p.
__device__ inline v16h load_b16(const _Float16* p) {
  v16h b;
  *(v8h*)&b       = *(const v8h*)(p);
  *((v8h*)&b + 1) = *(const v8h*)(p + 8);
  return b;
}

__device__ inline float half_reduce_max(float v) {
  v = fmaxf(v, __shfl_xor(v, 1));
  v = fmaxf(v, __shfl_xor(v, 2));
  v = fmaxf(v, __shfl_xor(v, 4));
  v = fmaxf(v, __shfl_xor(v, 8));
  return v;
}
__device__ inline float half_reduce_sum(float v) {
  v += __shfl_xor(v, 1);
  v += __shfl_xor(v, 2);
  v += __shfl_xor(v, 4);
  v += __shfl_xor(v, 8);
  return v;
}

// ---------------------------------------------------------------------------
__global__ __launch_bounds__(256) void cvt_f32_f16(const float* __restrict__ in,
                                                   _Float16* __restrict__ out, int n) {
  int i = blockIdx.x * 256 + threadIdx.x;
  if (i < n) out[i] = (_Float16)in[i];
}

// rel_proj[i] = dot(rel_pos[i,:], rpe_w) ; i in [0, 4095)
__global__ __launch_bounds__(256) void rpe_proj(const float* __restrict__ rel_pos,
                                                const float* __restrict__ rpe_w,
                                                float* __restrict__ relp) {
  int i = blockIdx.x * 256 + threadIdx.x;
  if (i < 4095) {
    float s = 0.f;
    #pragma unroll 8
    for (int d = 0; d < 64; ++d) s += rel_pos[i * 64 + d] * rpe_w[d];
    relp[i] = s;
  }
}

// ---------------------------------------------------------------------------
// qkv = xh[4096,1024] @ wh^T + qkv_b ; one wave per 64x64 tile (16 WMMAs/k-step).
// Scatter f16 into Q,K [b,h,t,hd] and V transposed [b,h,hd,t].
__global__ __launch_bounds__(32) void qkv_gemm(const _Float16* __restrict__ xh,
                                               const _Float16* __restrict__ wh,
                                               const float* __restrict__ qkv_b,
                                               _Float16* __restrict__ qh,
                                               _Float16* __restrict__ kh,
                                               _Float16* __restrict__ vth) {
  const int l    = threadIdx.x;
  const int nn   = l & 15;
  const int hi   = l >> 4;
  const int m0   = blockIdx.y * 64;   // 4096/64
  const int nb   = blockIdx.x * 64;   // 3072/64
  const int ksel = hi * 8;

  v8f acc[4][4] = {};

  for (int k0 = 0; k0 < 1024; k0 += 32) {
    v16h a[4], bm[4];
    #pragma unroll
    for (int s = 0; s < 4; ++s)
      a[s] = load_a16(xh + (size_t)(m0 + s * 16 + nn) * 1024 + k0, ksel);
    #pragma unroll
    for (int t = 0; t < 4; ++t)
      bm[t] = load_b16(wh + (size_t)(nb + t * 16 + nn) * 1024 + k0 + hi * 16);
    #pragma unroll
    for (int s = 0; s < 4; ++s)
      #pragma unroll
      for (int t = 0; t < 4; ++t)
        acc[s][t] = WMMA_F16(a[s], bm[t], acc[s][t]);
  }

  #pragma unroll
  for (int t = 0; t < 4; ++t) {
    const int n     = nb + t * 16 + nn;
    const float bia = qkv_b[n];
    const int which = n >> 10;      // 0=q 1=k 2=v
    const int d     = n & 1023;
    const int h     = d >> 6;
    const int hd    = d & 63;
    #pragma unroll
    for (int s = 0; s < 4; ++s) {
      #pragma unroll
      for (int r = 0; r < 8; ++r) {
        const int m  = m0 + s * 16 + hi * 8 + r;
        const int bb = m >> 11;
        const int tt = m & 2047;
        _Float16 val = (_Float16)(acc[s][t][r] + bia);
        if (which == 0)      qh [((bb * 16 + h) * 2048 + tt) * 64 + hd] = val;
        else if (which == 1) kh [((bb * 16 + h) * 2048 + tt) * 64 + hd] = val;
        else                 vth[((bb * 16 + h) * 64 + hd) * 2048 + tt] = val;
      }
    }
  }
}

// ---------------------------------------------------------------------------
// Flash attention: one wave per (b, h, 32-query tile). K B-tiles shared by the
// two 16-query sub-tiles; V tiles prefetched under the softmax VALU work.
__global__ __launch_bounds__(32) void attn_kernel(const _Float16* __restrict__ qh,
                                                  const _Float16* __restrict__ kh,
                                                  const _Float16* __restrict__ vth,
                                                  const float* __restrict__ relp,
                                                  _Float16* __restrict__ ah) {
  const int l  = threadIdx.x;
  const int nn = l & 15;
  const int hi = l >> 4;
  const int h  = blockIdx.y;   // 0..15
  const int b  = blockIdx.z;   // 0..1
  const int q0 = blockIdx.x * 32;  // 0..63 tiles of 32 queries

  const _Float16* qp = qh  + (size_t)((b * 16 + h) * 2048) * 64;
  const _Float16* kp = kh  + (size_t)((b * 16 + h) * 2048) * 64;
  const _Float16* vp = vth + (size_t)((b * 16 + h) * 64) * 2048;

  __shared__ __align__(16) _Float16 pbuf[32 * 32];

  const int ksel = hi * 8;
  v16h qa[2][2];  // [qsub][k-half]
  #pragma unroll
  for (int s = 0; s < 2; ++s) {
    qa[s][0] = load_a16(qp + (size_t)(q0 + s * 16 + nn) * 64 + 0,  ksel);
    qa[s][1] = load_a16(qp + (size_t)(q0 + s * 16 + nn) * 64 + 32, ksel);
  }

  v8f o[2][4] = {};
  float rM[2][8], rL[2][8];
  #pragma unroll
  for (int s = 0; s < 2; ++s)
    #pragma unroll
    for (int r = 0; r < 8; ++r) { rM[s][r] = -1e30f; rL[s][r] = 0.f; }
  const float scale = 0.125f;  // 64^-0.5

  for (int kc = 0; kc < 2048; kc += 32) {
    // S = Q @ K^T : 4 shared K B-tiles, 8 WMMAs (2 qsubs x 2 ktiles x 2 khalf)
    v8f sc[2][2] = {};  // [qsub][ktile]
    #pragma unroll
    for (int j = 0; j < 2; ++j) {
      #pragma unroll
      for (int kh2 = 0; kh2 < 2; ++kh2) {
        v16h bk = load_b16(kp + (size_t)(kc + j * 16 + nn) * 64 + kh2 * 32 + hi * 16);
        #pragma unroll
        for (int s = 0; s < 2; ++s)
          sc[s][j] = WMMA_F16(qa[s][kh2], bk, sc[s][j]);
      }
    }

    // Prefetch V tiles now; latency hides under the softmax below.
    v16h bv[4];
    #pragma unroll
    for (int t = 0; t < 4; ++t)
      bv[t] = load_b16(vp + (size_t)(t * 16 + nn) * 2048 + kc + hi * 16);

    // bias + online softmax (row = s*16 + hi*8 + r in C layout)
    #pragma unroll
    for (int s = 0; s < 2; ++s) {
      float p0[8], p1[8], cf[8];
      #pragma unroll
      for (int r = 0; r < 8; ++r) {
        const int qg = q0 + s * 16 + hi * 8 + r;
        float a0 = sc[s][0][r] * scale + relp[qg - (kc +      nn) + 2047];
        float a1 = sc[s][1][r] * scale + relp[qg - (kc + 16 + nn) + 2047];
        float mx = half_reduce_max(fmaxf(a0, a1));
        float nm = fmaxf(rM[s][r], mx);
        cf[r] = __expf(rM[s][r] - nm);
        rM[s][r] = nm;
        p0[r] = __expf(a0 - nm);
        p1[r] = __expf(a1 - nm);
        rL[s][r] = rL[s][r] * cf[r] + half_reduce_sum(p0[r] + p1[r]);
      }
      #pragma unroll
      for (int r = 0; r < 8; ++r) {
        #pragma unroll
        for (int t = 0; t < 4; ++t) o[s][t][r] *= cf[r];
        pbuf[(s * 16 + hi * 8 + r) * 32 +      nn] = (_Float16)p0[r];
        pbuf[(s * 16 + hi * 8 + r) * 32 + 16 + nn] = (_Float16)p1[r];
      }
    }
    __syncthreads();
    v16h pa[2];
    #pragma unroll
    for (int s = 0; s < 2; ++s)
      pa[s] = load_a16(&pbuf[(s * 16 + nn) * 32], ksel);
    __syncthreads();

    // O += P @ V : 8 WMMAs reusing the 4 prefetched V tiles
    #pragma unroll
    for (int s = 0; s < 2; ++s)
      #pragma unroll
      for (int t = 0; t < 4; ++t)
        o[s][t] = WMMA_F16(pa[s], bv[t], o[s][t]);
  }

  // normalize and store f16 into [b*T + t, h*64 + hd]
  #pragma unroll
  for (int s = 0; s < 2; ++s) {
    const int row0 = b * 2048 + q0 + s * 16 + hi * 8;
    #pragma unroll
    for (int r = 0; r < 8; ++r) {
      const float inv = 1.0f / rL[s][r];
      _Float16* dst = ah + (size_t)(row0 + r) * 1024 + h * 64 + nn;
      #pragma unroll
      for (int t = 0; t < 4; ++t)
        dst[t * 16] = (_Float16)(o[s][t][r] * inv);
    }
  }
}

// ---------------------------------------------------------------------------
// out = ah[4096,1024] @ out_w^T + out_b ; one wave per 64x64 tile.
__global__ __launch_bounds__(32) void out_gemm(const _Float16* __restrict__ ah,
                                               const _Float16* __restrict__ owh,
                                               const float* __restrict__ out_b,
                                               float* __restrict__ out) {
  const int l    = threadIdx.x;
  const int nn   = l & 15;
  const int hi   = l >> 4;
  const int m0   = blockIdx.y * 64;   // 4096/64
  const int nb   = blockIdx.x * 64;   // 1024/64
  const int ksel = hi * 8;

  v8f acc[4][4] = {};

  for (int k0 = 0; k0 < 1024; k0 += 32) {
    v16h a[4], bm[4];
    #pragma unroll
    for (int s = 0; s < 4; ++s)
      a[s] = load_a16(ah + (size_t)(m0 + s * 16 + nn) * 1024 + k0, ksel);
    #pragma unroll
    for (int t = 0; t < 4; ++t)
      bm[t] = load_b16(owh + (size_t)(nb + t * 16 + nn) * 1024 + k0 + hi * 16);
    #pragma unroll
    for (int s = 0; s < 4; ++s)
      #pragma unroll
      for (int t = 0; t < 4; ++t)
        acc[s][t] = WMMA_F16(a[s], bm[t], acc[s][t]);
  }

  #pragma unroll
  for (int t = 0; t < 4; ++t) {
    const int n     = nb + t * 16 + nn;
    const float bia = out_b[n];
    #pragma unroll
    for (int s = 0; s < 4; ++s)
      #pragma unroll
      for (int r = 0; r < 8; ++r)
        out[(size_t)(m0 + s * 16 + hi * 8 + r) * 1024 + n] = acc[s][t][r] + bia;
  }
}

// ---------------------------------------------------------------------------
extern "C" void kernel_launch(void* const* d_in, const int* in_sizes, int n_in,
                              void* d_out, int out_size, void* d_ws, size_t ws_size,
                              hipStream_t stream) {
  const float* x       = (const float*)d_in[0];  // [2,2048,1024]
  const float* qkv_w   = (const float*)d_in[1];  // [3072,1024]
  const float* qkv_b   = (const float*)d_in[2];  // [3072]
  const float* out_w   = (const float*)d_in[3];  // [1024,1024]
  const float* out_b   = (const float*)d_in[4];  // [1024]
  const float* rel_pos = (const float*)d_in[5];  // [4095,64]
  const float* rpe_w   = (const float*)d_in[6];  // [64]
  float* out = (float*)d_out;                    // [4096,1024]

  char* ws = (char*)d_ws;
  const size_t XH   = 0;                        // 4096*1024 f16 = 8 MB
  const size_t WH   = XH   + (size_t)8388608;   // 3072*1024 f16 = 6 MB
  const size_t OWH  = WH   + (size_t)6291456;   // 1024*1024 f16 = 2 MB
  const size_t QH   = OWH  + (size_t)2097152;   // 8 MB
  const size_t KH   = QH   + (size_t)8388608;   // 8 MB
  const size_t VTH  = KH   + (size_t)8388608;   // 8 MB (V transposed)
  const size_t AH   = VTH  + (size_t)8388608;   // 8 MB
  const size_t RELP = AH   + (size_t)8388608;   // 4095 f32

  _Float16* xh   = (_Float16*)(ws + XH);
  _Float16* wh   = (_Float16*)(ws + WH);
  _Float16* owh  = (_Float16*)(ws + OWH);
  _Float16* qh   = (_Float16*)(ws + QH);
  _Float16* kh   = (_Float16*)(ws + KH);
  _Float16* vth  = (_Float16*)(ws + VTH);
  _Float16* ah   = (_Float16*)(ws + AH);
  float*    relp = (float*)(ws + RELP);

  cvt_f32_f16<<<(4194304 + 255) / 256, 256, 0, stream>>>(x, xh, 4194304);
  cvt_f32_f16<<<(3145728 + 255) / 256, 256, 0, stream>>>(qkv_w, wh, 3145728);
  cvt_f32_f16<<<(1048576 + 255) / 256, 256, 0, stream>>>(out_w, owh, 1048576);
  rpe_proj<<<(4095 + 255) / 256, 256, 0, stream>>>(rel_pos, rpe_w, relp);

  qkv_gemm<<<dim3(48, 64), 32, 0, stream>>>(xh, wh, qkv_b, qh, kh, vth);
  attn_kernel<<<dim3(64, 16, 2), 32, 0, stream>>>(qh, kh, vth, relp, ah);
  out_gemm<<<dim3(16, 64), 32, 0, stream>>>(ah, owh, out_b, out);
}
